// Head_43843026158045
// MI455X (gfx1250) — matrix operand order repaired and checked by reading, not compiled, not run on hardware
//
#include <hip/hip_runtime.h>

typedef _Float16 f16;
typedef __attribute__((ext_vector_type(16))) _Float16 v16h;
typedef __attribute__((ext_vector_type(8)))  float    v8f;

#define B_ 8
#define T_ 2048
#define C_ 1024
#define H_ 64

// A-matrix (16xK, 16-bit): lane = M | (khalf<<4); element e -> K index.
// For fixed lane: e=0..7 -> k = hi*8 + 0..7 (contiguous), e=8..15 -> +16.
__device__ __forceinline__ int a_kmap(int e, int hi) {
  int v = e >> 1;
  return ((v & 3) << 1) + (e & 1) + (hi << 3) + ((v & 4) << 2);
}
// B-matrix (Kx16, 16-bit): lane = N | (khalf<<4); element e -> K = hi*8 + (e&7) + (e>=8 ? 16 : 0).
__device__ __forceinline__ int b_kmap(int e, int hi) {
  return (hi << 3) + (e & 7) + ((e & 8) << 1);
}

__device__ __forceinline__ v8f wmma_f16(v16h a, v16h b, v8f c) {
  return __builtin_amdgcn_wmma_f32_16x16x32_f16(
      /*neg_a=*/false, a, /*neg_b=*/false, b,
      /*c_mod=*/(short)0, c, /*reuse_a=*/false, /*reuse_b=*/false);
}

// Async DMA: copy `n16` 16-byte chunks global->LDS per lane, no VGPR roundtrip.
// the offset immediate is added to both the LDS and the global address.
__device__ __forceinline__ void async_b128_row(unsigned lds, const f16* gptr, int n16) {
  unsigned long long ga = (unsigned long long)(uintptr_t)gptr;
  if (n16 >= 4) {
    asm volatile(
        "global_load_async_to_lds_b128 %0, %1, off\n\t"
        "global_load_async_to_lds_b128 %0, %1, off offset:16\n\t"
        "global_load_async_to_lds_b128 %0, %1, off offset:32\n\t"
        "global_load_async_to_lds_b128 %0, %1, off offset:48\n\t"
        :: "v"(lds), "v"(ga) : "memory");
  }
  if (n16 == 8) {
    asm volatile(
        "global_load_async_to_lds_b128 %0, %1, off offset:64\n\t"
        "global_load_async_to_lds_b128 %0, %1, off offset:80\n\t"
        "global_load_async_to_lds_b128 %0, %1, off offset:96\n\t"
        "global_load_async_to_lds_b128 %0, %1, off offset:112\n\t"
        :: "v"(lds), "v"(ga) : "memory");
  }
}
__device__ __forceinline__ void async_wait0() {
  asm volatile("s_wait_asynccnt 0x0" ::: "memory");
}

// ---------------------------------------------------------------------------
// Kernel 0: one-time W convert+transpose: f32 [C][H] -> f16 Wt[3][H][C]
// ---------------------------------------------------------------------------
__global__ __launch_bounds__(256) void wt_kernel(
    const float* __restrict__ Wq, const float* __restrict__ Wk,
    const float* __restrict__ Wv, f16* __restrict__ Wt)
{
  int i = blockIdx.x * 256 + threadIdx.x;   // 0 .. 3*C*H-1
  int m = i >> 16;                          // C*H = 65536
  int rem = i & 65535;
  int k = rem >> 6, n = rem & 63;           // coalesced read over n
  const float* Wsrc = (m == 0) ? Wq : (m == 1) ? Wk : Wv;
  Wt[((size_t)(m * H_ + n)) * C_ + k] = (f16)Wsrc[(size_t)k * H_ + n];
}

// ---------------------------------------------------------------------------
// Kernel 1: QKV projection  [B*T, C] x [C, H] -> f16 Q(scaled), K, Vt in ws.
// 256 threads = 8 waves; each block does 128 token rows; K-chunked by 64.
// W arrives pre-transposed f16 -> LDS staging is pure b128 copies.
// ---------------------------------------------------------------------------
__global__ __launch_bounds__(256) void qkv_kernel(
    const float* __restrict__ x, const f16* __restrict__ Wt,
    f16* __restrict__ Qo, f16* __restrict__ Ko, f16* __restrict__ Vt)
{
  __shared__ f16 sX [128][72];     // [row][k], padded (144B rows, 16B-mult)
  __shared__ f16 sWt[192][72];     // [(mat*64+n)][k] transposed, padded

  const int tid  = threadIdx.x;
  const int lane = tid & 31, wave = tid >> 5;
  const int hi = lane >> 4, lo = lane & 15;
  const int rowbase = blockIdx.x * 128;

  v8f acc[12] = {};  // [q|k|v][4 n-tiles] of 16x16 f32

  for (int kc = 0; kc < C_; kc += 64) {
    // stage x chunk: 128 rows x 64 k (2048 float4, 8/thread, coalesced) + cvt
    #pragma unroll
    for (int i = 0; i < 8; ++i) {
      int idx = tid + i * 256;
      int row = idx >> 4, c4 = idx & 15;
      const float4 v4 =
          *(const float4*)&x[(size_t)(rowbase + row) * C_ + kc + c4 * 4];
      f16* dst = &sX[row][c4 * 4];
      dst[0] = (f16)v4.x; dst[1] = (f16)v4.y; dst[2] = (f16)v4.z; dst[3] = (f16)v4.w;
    }
    // stage Wt chunk: 192 rows x 64 halves = 1536 uint4, 6/thread, no cvt
    #pragma unroll
    for (int i = 0; i < 6; ++i) {
      int idx = tid + i * 256;          // 0..1535
      int row = idx >> 3, q = idx & 7;  // row = mat*64+n, q = 16B chunk
      const uint4 v4 = *(const uint4*)&Wt[(size_t)row * C_ + kc + q * 8];
      *(uint4*)&sWt[row][q * 8] = v4;
    }
    __syncthreads();

    #pragma unroll
    for (int kk = 0; kk < 2; ++kk) {   // two K=32 WMMA steps per 64-chunk
      v16h afrag;
      #pragma unroll
      for (int e = 0; e < 16; ++e)
        afrag[e] = sX[wave * 16 + lo][kk * 32 + a_kmap(e, hi)];
      #pragma unroll
      for (int t = 0; t < 12; ++t) {
        const int mat = t >> 2, nt = t & 3;
        v16h bfrag;
        #pragma unroll
        for (int e = 0; e < 16; ++e)   // contiguous in k -> 2x ds_load_b128
          bfrag[e] = sWt[mat * 64 + nt * 16 + lo][kk * 32 + b_kmap(e, hi)];
        acc[t] = wmma_f16(afrag, bfrag, acc[t]);
      }
    }
    __syncthreads();
  }

  // write Q (pre-scaled by H^-0.5 = 1/8, exact in f16), K row-major,
  // V transposed to Vt[b][h][t] so attention staging is pure b128 copies.
  #pragma unroll
  for (int t = 0; t < 12; ++t) {
    const int mat = t >> 2, nt = t & 3;
    #pragma unroll
    for (int r = 0; r < 8; ++r) {
      int token = rowbase + wave * 16 + r + hi * 8;   // C/D layout: M = r + 8*hi
      int h = nt * 16 + lo;
      if (mat == 0)      Qo[(size_t)token * H_ + h] = (f16)(acc[t][r] * 0.125f);
      else if (mat == 1) Ko[(size_t)token * H_ + h] = (f16)acc[t][r];
      else {
        int bb = token >> 11, tl = token & (T_ - 1);
        Vt[((size_t)(bb * H_ + h)) * T_ + tl] = (f16)acc[t][r];
      }
    }
  }
}

// ---------------------------------------------------------------------------
// Kernel 2: causal flash attention, 32-key blocks.
// One wave per 16-query tile, 4 waves/block, per-wave LDS slabs, no
// cross-wave sync. K/V tiles staged with async global->LDS b128 DMA.
// ---------------------------------------------------------------------------
__global__ __launch_bounds__(128) void attn_kernel(
    const f16* __restrict__ Q, const f16* __restrict__ K,
    const f16* __restrict__ Vt, float* __restrict__ out)
{
  __shared__ f16   sK [4][32][80];  // K tile: [key][h]   (n=key, k=h contiguous)
  __shared__ f16   sVt[4][64][40];  // V tile transposed: [h][key]
  __shared__ f16   sP [4][16][32];  // probs (A of P x V)
  __shared__ float sS [4][16][32];  // raw scores
  __shared__ float sAl[4][16];      // per-row rescale alpha
  __shared__ float sL [4][16];      // per-row running sum

  const int tid  = threadIdx.x;
  const int lane = tid & 31, w = tid >> 5;
  const int hi = lane >> 4, lo = lane & 15;

  const int qtile = blockIdx.x * 4 + w;
  const int b  = qtile >> 7;        // T/16 = 128 tiles per batch
  const int qt = qtile & 127;
  const int q0 = qt * 16;

  // Q fragments (reused across all key blocks); Q already scaled by 1/8
  v16h aQ[2];
  {
    const f16* qbase = Q + (size_t)(b * T_ + q0 + lo) * H_;
    #pragma unroll
    for (int c = 0; c < 2; ++c)
      #pragma unroll
      for (int e = 0; e < 16; ++e)
        aQ[c][e] = qbase[c * 32 + a_kmap(e, hi)];
  }

  v8f accO[4] = {};
  float m_i = -__builtin_inff();
  float l_i = 0.0f;

  const f16* Kb  = K  + (size_t)b * T_ * H_;
  const f16* Vtb = Vt + (size_t)b * H_ * T_;

  const int nblk = (qt >> 1) + 1;   // 32-key blocks covering keys <= q0+15
  for (int blk = 0; blk < nblk; ++blk) {
    const int key0 = blk * 32;

    // async-DMA K tile: lane owns one key row (64 halves = 8 x b128)
    async_b128_row((unsigned)(uintptr_t)&sK[w][lane][0],
                   Kb + (size_t)(key0 + lane) * H_, 8);
    // async-DMA Vt tile: lane owns h rows lane and lane+32 (32 halves each)
    async_b128_row((unsigned)(uintptr_t)&sVt[w][lane][0],
                   Vtb + (size_t)lane * T_ + key0, 4);
    async_b128_row((unsigned)(uintptr_t)&sVt[w][lane + 32][0],
                   Vtb + (size_t)(lane + 32) * T_ + key0, 4);
    // prefetch next block's K row while we compute on this one
    if (blk + 1 < nblk)
      __builtin_prefetch(Kb + (size_t)(key0 + 32 + lane) * H_, 0, 3);
    async_wait0();

    // scores: (Q/8) x K^T  -> two 16x16 tiles (keys lo, 16+lo)
    v8f accS[2] = {};
    #pragma unroll
    for (int half = 0; half < 2; ++half)
      #pragma unroll
      for (int c = 0; c < 2; ++c) {
        v16h bK;
        #pragma unroll
        for (int e = 0; e < 16; ++e)   // contiguous in h -> 2x ds_load_b128
          bK[e] = sK[w][half * 16 + lo][c * 32 + b_kmap(e, hi)];
        accS[half] = wmma_f16(aQ[c], bK, accS[half]);
      }

    // scores -> LDS so each of lanes 0..15 owns one query row
    #pragma unroll
    for (int r = 0; r < 8; ++r) {
      sS[w][r + hi * 8][lo]      = accS[0][r];
      sS[w][r + hi * 8][16 + lo] = accS[1][r];
    }

    // online softmax update (causal mask inside block)
    if (lane < 16) {
      const int row = lane;
      const int q = q0 + row;
      float tmax = -__builtin_inff();
      float s[32];
      #pragma unroll
      for (int j = 0; j < 32; ++j) {
        s[j] = ((key0 + j) <= q) ? sS[w][row][j] : -__builtin_inff();
        tmax = fmaxf(tmax, s[j]);
      }
      const float newm  = fmaxf(m_i, tmax);
      const float alpha = __expf(m_i - newm);
      float psum = 0.0f;
      #pragma unroll
      for (int j = 0; j < 32; ++j) {
        float p = ((key0 + j) <= q) ? __expf(s[j] - newm) : 0.0f;
        psum += p;
        sP[w][row][j] = (f16)p;
      }
      l_i = l_i * alpha + psum;
      m_i = newm;
      sAl[w][row] = alpha;
    }

    // rescale output accumulators by alpha(row)
    #pragma unroll
    for (int nt = 0; nt < 4; ++nt)
      #pragma unroll
      for (int r = 0; r < 8; ++r)
        accO[nt][r] *= sAl[w][r + hi * 8];

    // P x V: full 16x32 A, 32x64 B (no padding waste)
    v16h aP;
    #pragma unroll
    for (int e = 0; e < 16; ++e)       // two contiguous 8-half runs -> b128
      aP[e] = sP[w][lo][a_kmap(e, hi)];
    #pragma unroll
    for (int nt = 0; nt < 4; ++nt) {
      v16h bV;
      #pragma unroll
      for (int e = 0; e < 16; ++e)     // contiguous in key -> 2x ds_load_b128
        bV[e] = sVt[w][nt * 16 + lo][b_kmap(e, hi)];
      accO[nt] = wmma_f16(aP, bV, accO[nt]);
    }
  }

  // finalize: divide by running sum, store f32
  if (lane < 16) sL[w][lane] = l_i;
  #pragma unroll
  for (int nt = 0; nt < 4; ++nt) {
    #pragma unroll
    for (int r = 0; r < 8; ++r) {
      float val = accO[nt][r] / sL[w][r + hi * 8];
      out[(size_t)(b * T_ + q0 + r + hi * 8) * H_ + nt * 16 + lo] = val;
    }
  }
}

// ---------------------------------------------------------------------------
extern "C" void kernel_launch(void* const* d_in, const int* in_sizes, int n_in,
                              void* d_out, int out_size, void* d_ws, size_t ws_size,
                              hipStream_t stream) {
  (void)in_sizes; (void)n_in; (void)out_size; (void)ws_size;
  const float* x  = (const float*)d_in[0];
  const float* Wq = (const float*)d_in[1];
  const float* Wk = (const float*)d_in[2];
  const float* Wv = (const float*)d_in[3];

  const size_t nQKV = (size_t)B_ * T_ * H_;     // 1,048,576 elems
  f16* Q  = (f16*)d_ws;                         // 2 MB
  f16* K  = Q + nQKV;                           // 2 MB
  f16* Vt = K + nQKV;                           // 2 MB  [b][h][t]
  f16* Wt = Vt + nQKV;                          // 384 KB [3][H][C]

  wt_kernel  <<<(3 * C_ * H_) / 256, 256, 0, stream>>>(Wq, Wk, Wv, Wt);
  qkv_kernel <<<(B_ * T_) / 128, 256, 0, stream>>>(x, Wt, Q, K, Vt);
  attn_kernel<<<(B_ * (T_ / 16)) / 4, 128, 0, stream>>>(Q, K, Vt, (float*)d_out);
}
